// BertCRF_85787676770988
// MI455X (gfx1250) — matrix operand order repaired and checked by reading, not compiled, not run on hardware
//
#include <hip/hip_runtime.h>
#include <hip/hip_bf16.h>

// ---------------- problem constants (match reference) ----------------
#define VOCAB 30522
#define DEMB  768
#define TAGS  9
#define BATCH 64
#define SEQ   512
#define NTOK  (BATCH * SEQ)      // 32768 tokens
#define TPAD  16                 // tags padded to WMMA N=16

typedef __attribute__((ext_vector_type(16))) _Float16 v16h;
typedef __attribute__((ext_vector_type(8)))  float    v8f;

// =====================================================================
// Kernel 1: fused  emb_table[words]  ->  GEMM (x @ fc_w^T + b)  ->
//           log_softmax over tags, written as (NTOK, 16) f32 scratch.
// One wave per 16-token tile; V_WMMA_F32_16X16X32_F16, K-loop D/32 = 24.
// Memory-bound: streams ~96MB of embedding rows once (table is L2-resident
// at 192MB L2) -> ~4us HBM floor @ 23.3 TB/s. GEMM FLOPs are negligible.
// Dead-column trick: C cols n=9..15 are never consumed, so padded lanes
// load a clamped fc_w row unconditionally (no EXEC masking in hot loop).
// =====================================================================
__global__ __launch_bounds__(256) void bertcrf_embed_gemm_lsm(
    const int*   __restrict__ words,
    const float* __restrict__ emb,      // (VOCAB, DEMB)
    const float* __restrict__ fc_w,     // (TAGS, DEMB)
    const float* __restrict__ fc_b,     // (TAGS)
    float*       __restrict__ logits)   // (NTOK, TPAD) scratch out
{
    const int lane = threadIdx.x & 31;
    const int wave = blockIdx.x * (blockDim.x >> 5) + (threadIdx.x >> 5);
    const int m0   = wave * 16;          // first token of this tile
    const int row  = lane & 15;          // M row (A) / N col (B) within tile
    const int hi   = lane >> 4;          // half-wave selector
    const int nrow = (row < TAGS) ? row : 0;   // clamped: junk goes to dead cols

    const int    w    = words[m0 + row];
    const float* arow = emb  + (size_t)w    * DEMB;  // gathered x row
    const float* brow = fc_w + (size_t)nrow * DEMB;

    // start pulling this token's embedding row toward the WGP caches
    __builtin_prefetch(arow, 0, 3);

    v8f c = {};
    #pragma unroll 2
    for (int k0 = 0; k0 < DEMB; k0 += 32) {
        // ---- A tile, 16x32 f16, ISA wave32 layout ----
        // lanes 0-15: K = k0+{0..7, 16..23}; lanes 16-31: K = k0+{8..15, 24..31}
        v16h a;
        {
            const float* pa = arow + k0 + hi * 8;
            float4 p0 = *(const float4*)(pa + 0);
            float4 p1 = *(const float4*)(pa + 4);
            float4 p2 = *(const float4*)(pa + 16);
            float4 p3 = *(const float4*)(pa + 20);
            a[0]=(_Float16)p0.x;  a[1]=(_Float16)p0.y;  a[2]=(_Float16)p0.z;  a[3]=(_Float16)p0.w;
            a[4]=(_Float16)p1.x;  a[5]=(_Float16)p1.y;  a[6]=(_Float16)p1.z;  a[7]=(_Float16)p1.w;
            a[8]=(_Float16)p2.x;  a[9]=(_Float16)p2.y;  a[10]=(_Float16)p2.z; a[11]=(_Float16)p2.w;
            a[12]=(_Float16)p3.x; a[13]=(_Float16)p3.y; a[14]=(_Float16)p3.z; a[15]=(_Float16)p3.w;
        }
        // ---- B tile, 32x16 f16: B[k][n] = fc_w[n][k] (cols 9..15 = junk) ----
        // lanes 0-15: K = k0+{0..7, 16..23}; lanes 16-31: K = k0+{8..15, 24..31}
        v16h b;
        {
            const float* pb = brow + k0 + hi * 8;
            float4 q0 = *(const float4*)(pb + 0);
            float4 q1 = *(const float4*)(pb + 4);
            float4 q2 = *(const float4*)(pb + 16);
            float4 q3 = *(const float4*)(pb + 20);
            b[0]=(_Float16)q0.x;  b[1]=(_Float16)q0.y;  b[2]=(_Float16)q0.z;  b[3]=(_Float16)q0.w;
            b[4]=(_Float16)q1.x;  b[5]=(_Float16)q1.y;  b[6]=(_Float16)q1.z;  b[7]=(_Float16)q1.w;
            b[8]=(_Float16)q2.x;  b[9]=(_Float16)q2.y;  b[10]=(_Float16)q2.z; b[11]=(_Float16)q2.w;
            b[12]=(_Float16)q3.x; b[13]=(_Float16)q3.y; b[14]=(_Float16)q3.z; b[15]=(_Float16)q3.w;
        }
        c = __builtin_amdgcn_wmma_f32_16x16x32_f16(
                false, a, false, b, (short)0, c, false, false);
    }

    // ---- epilogue: +bias, log_softmax over tags (16-lane half reductions) ----
    // C/D layout: VGPR r -> (M=r, N=lane) lanes 0-15 ; (M=r+8, N=lane-16) 16-31
    const float bias = (row < TAGS) ? fc_b[row] : 0.0f;
    #pragma unroll
    for (int r = 0; r < 8; ++r) {
        float v  = c[r] + bias;
        float vm = (row < TAGS) ? v : -3.0e38f;   // mask padded cols out
        float mx = vm;
        #pragma unroll
        for (int off = 1; off < 16; off <<= 1)
            mx = fmaxf(mx, __shfl_xor(mx, off, 32));
        float e  = (row < TAGS) ? __expf(v - mx) : 0.0f;
        float ss = e;
        #pragma unroll
        for (int off = 1; off < 16; off <<= 1)
            ss += __shfl_xor(ss, off, 32);
        float lz = mx + __logf(ss);
        logits[(size_t)(m0 + r + hi * 8) * TPAD + row] = v - lz;
    }
}

// =====================================================================
// Kernel 2: CRF forward scan + gold score. One wave per batch element;
// lanes 0..8 own states, alpha replicated per-lane via shuffles.
// All operands (2MB logits, words/target) are L2-resident after kernel 1.
// =====================================================================
__global__ __launch_bounds__(32) void bertcrf_scan(
    const float* __restrict__ logits,     // (NTOK, TPAD)
    const int*   __restrict__ words,
    const int*   __restrict__ target,
    const float* __restrict__ trans,      // (TAGS, TAGS)
    const float* __restrict__ start_s,    // (TAGS)
    const float* __restrict__ end_s,      // (TAGS)
    float*       __restrict__ out)        // (BATCH)
{
    const int b    = blockIdx.x;
    const int lane = threadIdx.x;
    const int base = b * SEQ;
    const bool act = lane < TAGS;

    // lane t' holds trans[j][t'] for all j
    float tcol[TAGS];
    #pragma unroll
    for (int j = 0; j < TAGS; ++j)
        tcol[j] = act ? trans[j * TAGS + lane] : 0.0f;

    // alpha0 = logits[s=0] + start ; replicate across lanes
    float myv = act ? (logits[(size_t)base * TPAD + lane] + start_s[lane]) : 0.0f;
    float a[TAGS];
    #pragma unroll
    for (int j = 0; j < TAGS; ++j) a[j] = __shfl(myv, j, 32);

    for (int s = 1; s < SEQ; ++s) {
        float emit = act ? logits[(size_t)(base + s) * TPAD + lane] : 0.0f;
        float mx = -3.0e38f;
        #pragma unroll
        for (int j = 0; j < TAGS; ++j) mx = fmaxf(mx, a[j] + tcol[j]);
        float ssum = 0.0f;
        #pragma unroll
        for (int j = 0; j < TAGS; ++j) ssum += __expf(a[j] + tcol[j] - mx);
        float newv = mx + __logf(ssum) + emit;
        float upd  = (words[base + s] != 0) ? newv : (act ? a[lane] : 0.0f);
        #pragma unroll
        for (int j = 0; j < TAGS; ++j) a[j] = __shfl(upd, j, 32);
    }

    // normalizer = logsumexp(alpha + end)
    float v  = act ? (a[lane] + end_s[lane]) : -3.0e38f;
    float mx = v;
    #pragma unroll
    for (int off = 1; off < 32; off <<= 1) mx = fmaxf(mx, __shfl_xor(mx, off, 32));
    float ss = act ? __expf(v - mx) : 0.0f;
    #pragma unroll
    for (int off = 1; off < 32; off <<= 1) ss += __shfl_xor(ss, off, 32);
    const float normalizer = mx + __logf(ss);

    // gold path score (lane-strided, wave-reduced)
    float gsum = 0.0f;
    int   cnt  = 0;
    for (int s = lane; s < SEQ; s += 32) {
        const int wv = words[base + s];
        const int tg = target[base + s];
        if (wv != 0) { gsum += logits[(size_t)(base + s) * TPAD + tg]; cnt++; }
        if (s + 1 < SEQ && words[base + s + 1] != 0)
            gsum += trans[tg * TAGS + target[base + s + 1]];
    }
    #pragma unroll
    for (int off = 1; off < 32; off <<= 1) {
        gsum += __shfl_xor(gsum, off, 32);
        cnt  += __shfl_xor(cnt,  off, 32);
    }
    if (lane == 0) {
        const int last_tag = target[base + (cnt - 1)];
        const float gold = gsum + start_s[target[base]] + end_s[last_tag];
        out[b] = normalizer - gold;
    }
}

extern "C" void kernel_launch(void* const* d_in, const int* in_sizes, int n_in,
                              void* d_out, int out_size, void* d_ws, size_t ws_size,
                              hipStream_t stream) {
    (void)in_sizes; (void)n_in; (void)out_size; (void)ws_size;
    const int*   words   = (const int*)  d_in[0];
    const int*   target  = (const int*)  d_in[1];
    const float* emb     = (const float*)d_in[2];
    const float* fc_w    = (const float*)d_in[3];
    const float* fc_b    = (const float*)d_in[4];
    const float* trans   = (const float*)d_in[5];
    const float* start_s = (const float*)d_in[6];
    const float* end_s   = (const float*)d_in[7];
    float* out    = (float*)d_out;
    float* logits = (float*)d_ws;        // (NTOK, 16) f32 = 2 MB scratch

    // 2048 16-token tiles, 8 waves/block (wave32) -> 256 blocks of 256 threads
    bertcrf_embed_gemm_lsm<<<NTOK / 16 / 8, 256, 0, stream>>>(
        words, emb, fc_w, fc_b, logits);
    // one wave per batch element
    bertcrf_scan<<<BATCH, 32, 0, stream>>>(
        logits, words, target, trans, start_s, end_s, out);
}
